// MultiHeadAttention_2671469658755
// MI455X (gfx1250) — matrix operand order repaired and checked by reading, not compiled, not run on hardware
//
#include <hip/hip_runtime.h>

#define BB 4
#define SS 2048
#define DD 128
#define HH 8
#define EE 1024           // HH*DD
#define S16 (SS/16)

typedef __attribute__((ext_vector_type(16))) __bf16 v16bf;
typedef __attribute__((ext_vector_type(8)))  float  v8f;

union ABits { unsigned int u[8]; v16bf v; };

// ---- f32 -> bf16 (round to nearest even) ----
__device__ inline unsigned short f2bf(float f) {
  union { float f; unsigned int u; } x; x.f = f;
  unsigned int r = x.u + 0x7FFFu + ((x.u >> 16) & 1u);
  return (unsigned short)(r >> 16);
}
__device__ inline unsigned int pack2bf(float lo, float hi) {
  return (unsigned int)f2bf(lo) | ((unsigned int)f2bf(hi) << 16);
}

// ---- WMMA wrapper: D = A(16x32 bf16) * B(32x16 bf16) + C(16x16 f32) ----
__device__ inline v8f wmma_bf16(v16bf a, v16bf b, v8f c) {
  return __builtin_amdgcn_wmma_f32_16x16x32_bf16(false, a, false, b, (short)0, c, false, false);
}

// A operand loader: source row-major, contraction contiguous within row.
// Per 16-bit A layout: lane L holds row M=L%16,, VGPR p holds K-pair at
// K = (p<4?0:16) + (lane>=16?8:0) + 2*(p&3).
__device__ inline v16bf load_a_row(const unsigned short* row, int kbase, int lane) {
  ABits a;
  const unsigned int* p32 = (const unsigned int*)(row + kbase);
  int hi4 = (lane & 16) ? 4 : 0;
#pragma unroll
  for (int p = 0; p < 8; ++p) {
    int dw = ((p & 4) ? 8 : 0) + hi4 + (p & 3);
    a.u[p] = p32[dw];
  }
  return a.v;
}

// B operand loader: memory layout [col][contraction] (contraction contiguous).
// Lane L<16 -> column col0+L, K=0..7,16..23 ; lane L+16 -> same column, K=8..15,24..31.
__device__ inline v16bf load_b_ccont(const unsigned short* base, long colStride,
                                     int col0, int kbase, int lane) {
  ABits b;
  const unsigned int* p32 = (const unsigned int*)
      (base + (long)(col0 + (lane & 15)) * colStride + kbase + ((lane & 16) ? 8 : 0));
#pragma unroll
  for (int p = 0; p < 8; ++p) b.u[p] = p32[((p & 4) ? 8 : 0) + (p & 3)];
  return b.v;
}

// ---------------- prep kernels ----------------
__global__ void k_cvt_bf16(const float* __restrict__ in, unsigned short* __restrict__ out, int n) {
  int i = blockIdx.x * blockDim.x + threadIdx.x;
  if (i < n) out[i] = f2bf(in[i]);
}
// out[c*rows + r] = in[r*cols + c]
__global__ void k_transpose_cvt(const float* __restrict__ in, unsigned short* __restrict__ out,
                                int rows, int cols) {
  int i = blockIdx.x * blockDim.x + threadIdx.x;
  if (i < rows * cols) {
    int r = i / cols, c = i % cols;
    out[(long)c * rows + r] = f2bf(in[(long)r * cols + c]);
  }
}

// ---------------- QKV projection ----------------
// grid: B*S/16 blocks, 256 threads (8 waves; wave = head)
__global__ __launch_bounds__(256) void k_qkv(
    const unsigned short* __restrict__ encb,
    const unsigned short* __restrict__ WqT,
    const unsigned short* __restrict__ WkT,
    const unsigned short* __restrict__ WvT,
    unsigned short* __restrict__ Qb,   // [b,h,s,d]
    unsigned short* __restrict__ Kb,   // [b,h,s,d]
    unsigned short* __restrict__ Vt) { // [b,h,d,s]
  int blk = blockIdx.x;
  int b = blk / S16, st = blk % S16;
  int qbase = st * 16;
  int wave = threadIdx.x >> 5;
  int lane = threadIdx.x & 31;
  int h = wave;
  int hi8 = (lane & 16) ? 8 : 0;

  const unsigned short* arow = encb + (long)(b * SS + qbase + (lane & 15)) * DD;
  v16bf A[4];
#pragma unroll
  for (int kk = 0; kk < 4; ++kk) A[kk] = load_a_row(arow, kk * 32, lane);

  const unsigned short* Ws[3] = { WqT, WkT, WvT };
  for (int m = 0; m < 3; ++m) {
    const unsigned short* WT = Ws[m];
#pragma unroll
    for (int nt = 0; nt < 8; ++nt) {
      v8f c = {0.f,0.f,0.f,0.f,0.f,0.f,0.f,0.f};
#pragma unroll
      for (int kk = 0; kk < 4; ++kk) {
        v16bf Bm = load_b_ccont(WT, DD, h * DD + nt * 16, kk * 32, lane);
        c = wmma_bf16(A[kk], Bm, c);
      }
      int dcol = nt * 16 + (lane & 15);
      if (m < 2) {
        unsigned short* dst = (m == 0) ? Qb : Kb;
#pragma unroll
        for (int r = 0; r < 8; ++r) {
          int srow = qbase + hi8 + r;
          dst[((long)(b * HH + h) * SS + srow) * DD + dcol] = f2bf(c[r]);
        }
      } else {
        // V transposed: lane owns column dcol, 8 consecutive s positions
        unsigned int* dst = (unsigned int*)
            (Vt + ((long)(b * HH + h) * DD + dcol) * SS + qbase + hi8);
#pragma unroll
        for (int r = 0; r < 4; ++r) dst[r] = pack2bf(c[2*r], c[2*r+1]);
      }
    }
  }
}

// ---------------- fused attention ----------------
// grid: B*H*S/16 blocks, 128 threads (4 waves). 128KB LDS score strip.
__global__ __launch_bounds__(128) void k_attn(
    const unsigned short* __restrict__ Qb,
    const unsigned short* __restrict__ Kb,
    const unsigned short* __restrict__ Vt,
    const int* __restrict__ mask,
    float* __restrict__ attn_out,        // [b,h,s,s]
    unsigned short* __restrict__ Ocat) { // [b,s,e] bf16
  __shared__ float sc[16][SS];
  int blk = blockIdx.x;
  int b   = blk / (HH * S16);
  int rem = blk % (HH * S16);
  int h   = rem / S16;
  int qt  = rem % S16;
  int qbase = qt * 16;
  int wave = threadIdx.x >> 5;
  int lane = threadIdx.x & 31;
  int hi8 = (lane & 16) ? 8 : 0;
  long bh = (long)(b * HH + h);

  // ---- phase 1: scores = Q @ K^T ----
  const unsigned short* qrow = Qb + (bh * SS + qbase + (lane & 15)) * DD;
  v16bf A[4];
#pragma unroll
  for (int kk = 0; kk < 4; ++kk) A[kk] = load_a_row(qrow, kk * 32, lane);
  const unsigned short* Kbase = Kb + bh * SS * DD;
  for (int kt = wave * 32; kt < wave * 32 + 32; ++kt) {
    v8f c = {0.f,0.f,0.f,0.f,0.f,0.f,0.f,0.f};
#pragma unroll
    for (int kk = 0; kk < 4; ++kk) {
      v16bf Bm = load_b_ccont(Kbase, DD, kt * 16, kk * 32, lane);
      c = wmma_bf16(A[kk], Bm, c);
    }
    int col = kt * 16 + (lane & 15);
#pragma unroll
    for (int r = 0; r < 8; ++r) sc[hi8 + r][col] = c[r];
  }
  __syncthreads();

  // ---- phase 2: mask + scale + softmax (8 threads per row) ----
  const float scale = 0.08838834764831845f; // 1/sqrt(128)
  int rrow = threadIdx.x >> 3;
  int sub  = threadIdx.x & 7;
  int q = qbase + rrow;
  const int* mrow = mask + ((long)b * SS + q) * SS;
  float m = -3.4e38f;
  for (int k = sub; k < SS; k += 8) {
    float v = sc[rrow][k] * scale;
    if (mrow[k] == 0) v = -1e9f;
    sc[rrow][k] = v;
    m = fmaxf(m, v);
  }
#pragma unroll
  for (int off = 4; off >= 1; off >>= 1) m = fmaxf(m, __shfl_xor(m, off, 8));
  float ssum = 0.f;
  for (int k = sub; k < SS; k += 8) ssum += __expf(sc[rrow][k] - m);
#pragma unroll
  for (int off = 4; off >= 1; off >>= 1) ssum += __shfl_xor(ssum, off, 8);
  float inv = 1.f / ssum;
  float* arow_out = attn_out + (bh * SS + q) * SS;
  for (int k = sub; k < SS; k += 8) {
    float p = __expf(sc[rrow][k] - m) * inv;
    sc[rrow][k] = p;
    arow_out[k] = p;          // attn written exactly once
  }
  __syncthreads();

  // ---- phase 3: O = attn @ V (each wave: 2 column tiles of 16) ----
  const unsigned short* Vbase = Vt + bh * DD * SS;
  v8f C[2];
  C[0] = (v8f){0.f,0.f,0.f,0.f,0.f,0.f,0.f,0.f};
  C[1] = (v8f){0.f,0.f,0.f,0.f,0.f,0.f,0.f,0.f};
  int arl = lane & 15;
  for (int kc = 0; kc < 64; ++kc) {
    ABits a;
#pragma unroll
    for (int p = 0; p < 8; ++p) {
      int k0 = kc * 32 + ((p & 4) ? 16 : 0) + hi8 + 2 * (p & 3);
      a.u[p] = pack2bf(sc[arl][k0], sc[arl][k0 + 1]);
    }
    v16bf B0 = load_b_ccont(Vbase, SS, (2 * wave) * 16, kc * 32, lane);
    v16bf B1 = load_b_ccont(Vbase, SS, (2 * wave + 1) * 16, kc * 32, lane);
    C[0] = wmma_bf16(a.v, B0, C[0]);
    C[1] = wmma_bf16(a.v, B1, C[1]);
  }
#pragma unroll
  for (int j = 0; j < 2; ++j) {
    int ecol = h * DD + (2 * wave + j) * 16 + (lane & 15);
#pragma unroll
    for (int r = 0; r < 8; ++r) {
      int srow = qbase + hi8 + r;
      Ocat[(long)(b * SS + srow) * EE + ecol] = f2bf(C[j][r]);
    }
  }
}

// ---------------- out-proj + residual + LayerNorm ----------------
// grid: B*S/16 blocks, 256 threads (8 waves; wave = 16-col tile)
__global__ __launch_bounds__(256) void k_oproj_ln(
    const unsigned short* __restrict__ Ocat,
    const unsigned short* __restrict__ WoT,  // [128][1024]
    const float* __restrict__ enc,
    const float* __restrict__ ln_scale,
    const float* __restrict__ ln_bias,
    float* __restrict__ out) {
  __shared__ float so[16][DD];
  int blk = blockIdx.x;
  int b = blk / S16, st = blk % S16;
  int qbase = st * 16;
  int wave = threadIdx.x >> 5, lane = threadIdx.x & 31;
  int hi8 = (lane & 16) ? 8 : 0;

  const unsigned short* arow = Ocat + (long)(b * SS + qbase + (lane & 15)) * EE;
  v8f c = {0.f,0.f,0.f,0.f,0.f,0.f,0.f,0.f};
  for (int kc = 0; kc < 32; ++kc) {
    v16bf a  = load_a_row(arow, kc * 32, lane);
    v16bf Bm = load_b_ccont(WoT, EE, wave * 16, kc * 32, lane);
    c = wmma_bf16(a, Bm, c);
  }
  int col = wave * 16 + (lane & 15);
#pragma unroll
  for (int r = 0; r < 8; ++r) {
    int srow = qbase + hi8 + r;
    so[hi8 + r][col] = c[r] + enc[(long)(b * SS + srow) * DD + col];
  }
  __syncthreads();

  // LayerNorm: 16 threads per row, 8 cols each
  int rrow = threadIdx.x >> 4;
  int sub  = threadIdx.x & 15;
  float s = 0.f;
  for (int ci = sub; ci < DD; ci += 16) s += so[rrow][ci];
#pragma unroll
  for (int off = 8; off >= 1; off >>= 1) s += __shfl_xor(s, off, 16);
  float mean = s * (1.0f / DD);
  float v2 = 0.f;
  for (int ci = sub; ci < DD; ci += 16) { float d = so[rrow][ci] - mean; v2 += d * d; }
#pragma unroll
  for (int off = 8; off >= 1; off >>= 1) v2 += __shfl_xor(v2, off, 16);
  float rstd = rsqrtf(v2 * (1.0f / DD) + 1e-6f);
  float* orow = out + (long)(b * SS + qbase + rrow) * DD;
  for (int ci = sub; ci < DD; ci += 16)
    orow[ci] = (so[rrow][ci] - mean) * rstd * ln_scale[ci] + ln_bias[ci];
}

extern "C" void kernel_launch(void* const* d_in, const int* in_sizes, int n_in,
                              void* d_out, int out_size, void* d_ws, size_t ws_size,
                              hipStream_t stream) {
  (void)in_sizes; (void)n_in; (void)out_size; (void)ws_size;
  const float* enc  = (const float*)d_in[0];
  const int*   mask = (const int*)  d_in[1];
  const float* Wq   = (const float*)d_in[2];
  const float* Wk   = (const float*)d_in[3];
  const float* Wv   = (const float*)d_in[4];
  const float* Wo   = (const float*)d_in[5];
  const float* lns  = (const float*)d_in[6];
  const float* lnb  = (const float*)d_in[7];

  float* out  = (float*)d_out;
  float* attn = out + (long)BB * SS * DD;   // outputs concatenated: (out, attn)

  char* ws = (char*)d_ws;
  size_t off = 0;
  auto alloc = [&](size_t bytes) -> char* {
    char* p = ws + off; off += (bytes + 255) & ~(size_t)255; return p;
  };
  unsigned short* encb = (unsigned short*)alloc((size_t)BB * SS * DD * 2);
  unsigned short* WqT  = (unsigned short*)alloc((size_t)EE * DD * 2);
  unsigned short* WkT  = (unsigned short*)alloc((size_t)EE * DD * 2);
  unsigned short* WvT  = (unsigned short*)alloc((size_t)EE * DD * 2);
  unsigned short* WoT  = (unsigned short*)alloc((size_t)DD * EE * 2);
  unsigned short* Qb   = (unsigned short*)alloc((size_t)BB * HH * SS * DD * 2);
  unsigned short* Kb   = (unsigned short*)alloc((size_t)BB * HH * SS * DD * 2);
  unsigned short* Vt   = (unsigned short*)alloc((size_t)BB * HH * DD * SS * 2);
  unsigned short* Ocat = (unsigned short*)alloc((size_t)BB * SS * EE * 2);

  int n1 = BB * SS * DD;
  k_cvt_bf16<<<(n1 + 255) / 256, 256, 0, stream>>>(enc, encb, n1);
  int nw = DD * EE;
  k_transpose_cvt<<<(nw + 255) / 256, 256, 0, stream>>>(Wq, WqT, DD, EE);
  k_transpose_cvt<<<(nw + 255) / 256, 256, 0, stream>>>(Wk, WkT, DD, EE);
  k_transpose_cvt<<<(nw + 255) / 256, 256, 0, stream>>>(Wv, WvT, DD, EE);
  k_transpose_cvt<<<(nw + 255) / 256, 256, 0, stream>>>(Wo, WoT, EE, DD);

  k_qkv<<<BB * S16, 256, 0, stream>>>(encb, WqT, WkT, WvT, Qb, Kb, Vt);
  k_attn<<<BB * HH * S16, 128, 0, stream>>>(Qb, Kb, Vt, mask, attn, Ocat);
  k_oproj_ln<<<BB * S16, 256, 0, stream>>>(Ocat, WoT, enc, lns, lnb, out);
}
